// CausalSelfAttention_40973988004373
// MI455X (gfx1250) — matrix operand order repaired and checked by reading, not compiled, not run on hardware
//
#include <hip/hip_runtime.h>

// CDNA5 / gfx1250 causal self-attention: bf16 WMMA + async global->LDS staging.
// B=16, T=1024, C=768, H=12, d=64.

typedef __bf16 bf16;
typedef __attribute__((ext_vector_type(16))) bf16 v16bf;
typedef __attribute__((ext_vector_type(8)))  bf16 v8bf;
typedef __attribute__((ext_vector_type(8)))  float v8f;

union ABfrag { v16bf v; struct { v8bf lo, hi; } h; };

__device__ __forceinline__ bf16 f2bf(float f) {
  unsigned u = __builtin_bit_cast(unsigned, f);
  unsigned r = u + 0x7FFFu + ((u >> 16) & 1u);   // round-to-nearest-even
  unsigned short hv = (unsigned short)(r >> 16);
  return __builtin_bit_cast(bf16, hv);
}

__device__ __forceinline__ v8f wmma_bf16(v16bf a, v16bf b, v8f c) {
  return __builtin_amdgcn_wmma_f32_16x16x32_bf16(false, a, false, b, (short)0, c,
                                                 false, false);
}

// Per-lane async copy of 16 bytes global -> LDS (ASYNCcnt-tracked).
__device__ __forceinline__ void async_ld16(const bf16* g, bf16* lds) {
  asm volatile("global_load_async_to_lds_b128 %0, %1, off"
               :: "v"((unsigned)(size_t)lds),
                  "v"((unsigned long long)(size_t)g)
               : "memory");
}
__device__ __forceinline__ void wait_async0() {
  asm volatile("s_wait_asynccnt 0" ::: "memory");
}

// ---------------- precision-conversion kernels ----------------

__global__ void cvt_f32_bf16(const float* __restrict__ src, bf16* __restrict__ dst, int n) {
  int i = blockIdx.x * blockDim.x + threadIdx.x;
  if (i < n) dst[i] = f2bf(src[i]);
}

// W is [K][N] fp32 -> WT is [N][K] bf16
__global__ void transpose_to_bf16(const float* __restrict__ W, bf16* __restrict__ WT,
                                  int K, int N) {
  int i = blockIdx.x * blockDim.x + threadIdx.x;
  if (i >= K * N) return;
  int n = i / K, k = i - n * K;
  WT[i] = f2bf(W[(size_t)k * N + n]);
}

// ---------------- shared GEMM: C = A[16384x768] @ WT^T + bias ----------------
// Block: 256 threads (8 waves), tile 128(M) x 128(N), K-step 32, double-buffered
// LDS staged via async global->LDS. Wave = 32(M) x 64(N) strip (8 accumulators).
// mode 0: scatter q/k/v (bf16) into Qb / Kb / Vt[b,h,d,T].  mode 1: fp32 out.
__global__ __launch_bounds__(256) void gemm_wmma(
    const bf16* __restrict__ A, const bf16* __restrict__ WT,
    const float* __restrict__ bias, int N, int mode,
    bf16* __restrict__ Qb, bf16* __restrict__ Kb, bf16* __restrict__ Vt,
    float* __restrict__ outf) {
  __shared__ bf16 As[2][128 * 32];
  __shared__ bf16 Bs[2][128 * 32];
  const int tid = threadIdx.x;
  const int nblk = N >> 7;
  const int mt = blockIdx.x / nblk, nt = blockIdx.x % nblk;
  const int mbase = mt << 7, nbblk = nt << 7;
  const int lane = tid & 31, wv = tid >> 5;
  const int wm = wv & 3, wn = wv >> 1 & 2 ? 0 : 0; // placeholder, computed below
  const int wrow = (wv & 3) * 32;        // wave M offset in block tile
  const int wcol = (wv >> 2) * 64;       // wave N offset in block tile
  const int ln = lane & 15, g = lane >> 4;
  (void)wm; (void)wn;

  const bf16* Ag = A + (size_t)mbase * 768;
  const bf16* Bg = WT + (size_t)nbblk * 768;

  // Stage one 128x32 tile of A and B (512 x 16B chunks, 2 per thread each).
  auto stage = [&](int bi, int k0) {
#pragma unroll
    for (int i = 0; i < 2; ++i) {
      const int idx = tid + i * 256;
      const int row = idx >> 2, ch = (idx & 3) * 8;
      async_ld16(Ag + (size_t)row * 768 + k0 + ch, &As[bi][row * 32 + ch]);
      async_ld16(Bg + (size_t)row * 768 + k0 + ch, &Bs[bi][row * 32 + ch]);
    }
  };

  v8f acc[2][4];
#pragma unroll
  for (int s = 0; s < 2; ++s)
#pragma unroll
    for (int n = 0; n < 4; ++n) acc[s][n] = 0.f;

  stage(0, 0);
  wait_async0();
  __syncthreads();

  for (int ks = 0; ks < 24; ++ks) {
    const int bi = ks & 1;
    if (ks + 1 < 24) stage(bi ^ 1, (ks + 1) * 32);   // async prefetch next K-slab

    ABfrag a[2];
#pragma unroll
    for (int s = 0; s < 2; ++s) {
      const bf16* ap = &As[bi][(wrow + s * 16 + ln) * 32];
      a[s].h.lo = *(const v8bf*)(ap + 16 * g);
      a[s].h.hi = *(const v8bf*)(ap + 8 + 16 * g);
    }
#pragma unroll
    for (int n = 0; n < 4; ++n) {
      ABfrag b;
      b.v = *(const v16bf*)(&Bs[bi][(wcol + n * 16 + ln) * 32 + 16 * g]);
      acc[0][n] = wmma_bf16(a[0].v, b.v, acc[0][n]);
      acc[1][n] = wmma_bf16(a[1].v, b.v, acc[1][n]);
    }
    wait_async0();        // own async writes for buffer bi^1 have landed
    __syncthreads();      // everyone done reading bi, everyone's stage landed
  }

  const int nbase = nbblk + wcol;        // 64-aligned
  if (mode == 0) {
    const int which = nbase / 768;
    const int h = (nbase % 768) / 64;
#pragma unroll
    for (int n = 0; n < 4; ++n) {
      const float bv = bias[nbase + n * 16 + ln];
      const int d = n * 16 + ln;
#pragma unroll
      for (int s = 0; s < 2; ++s) {
#pragma unroll
        for (int r = 0; r < 8; ++r) {
          const int m = mbase + wrow + s * 16 + r + 8 * g;
          const int bi2 = m >> 10, t = m & 1023;
          const bf16 o = f2bf(acc[s][n][r] + bv);
          const size_t bh = (size_t)bi2 * 12 + h;
          if (which == 0)      Qb[(bh * 1024 + t) * 64 + d] = o;
          else if (which == 1) Kb[(bh * 1024 + t) * 64 + d] = o;
          else                 Vt[(bh * 64 + d) * 1024 + t] = o;
        }
      }
    }
  } else {
#pragma unroll
    for (int n = 0; n < 4; ++n) {
      const int col = nbase + n * 16 + ln;
      const float bv = bias[col];
#pragma unroll
      for (int s = 0; s < 2; ++s) {
#pragma unroll
        for (int r = 0; r < 8; ++r) {
          const int m = mbase + wrow + s * 16 + r + 8 * g;
          outf[(size_t)m * 768 + col] = acc[s][n][r] + bv;
        }
      }
    }
  }
}

// ---------------- flash attention: one wave per (b,h, 32-query block) ----------------
__global__ __launch_bounds__(128) void attn_wmma(
    const bf16* __restrict__ Qb, const bf16* __restrict__ Kb,
    const bf16* __restrict__ Vt, bf16* __restrict__ ybf) {
  __shared__ bf16 plds[4][1024];          // per-wave 2x(16x32) P staging
  const int lane = threadIdx.x & 31, wv = threadIdx.x >> 5;
  const int gw = blockIdx.x * 4 + wv;
  const int q32 = gw & 31;                // 32-query block [0,32)
  const int bh  = gw >> 5;                // b*12+h in [0,192)
  const int ln = lane & 15, g = lane >> 4;
  const int qbase = q32 * 32;
  const bf16* Qp = Qb + (size_t)bh * 65536;
  const bf16* Kp = Kb + (size_t)bh * 65536;
  const bf16* Vp = Vt + (size_t)bh * 65536;
  bf16* pbuf = plds[wv];

  ABfrag qf[2][2];                        // [m-tile][k-step]
#pragma unroll
  for (int s = 0; s < 2; ++s) {
    const int qrow = qbase + s * 16 + ln;
#pragma unroll
    for (int ks = 0; ks < 2; ++ks) {
      qf[s][ks].h.lo = *(const v8bf*)(Qp + qrow * 64 + ks * 32 + 16 * g);
      qf[s][ks].h.hi = *(const v8bf*)(Qp + qrow * 64 + ks * 32 + 8 + 16 * g);
    }
  }

  v8f acc[2][4];
  float m[2][8], l[2][8];
#pragma unroll
  for (int s = 0; s < 2; ++s) {
#pragma unroll
    for (int n = 0; n < 4; ++n) acc[s][n] = 0.f;
#pragma unroll
    for (int r = 0; r < 8; ++r) { m[s][r] = -1e30f; l[s][r] = 0.f; }
  }

  const int kbmax = q32;                  // (qbase+31)>>5
  for (int kb = 0; kb <= kbmax; ++kb) {
    const int k0 = kb * 32;
    if (kb < kbmax) {                     // prefetch next key/value block
      __builtin_prefetch(Kp + (k0 + 32 + ln) * 64, 0, 0);
      __builtin_prefetch(Vp + (size_t)ln * 1024 + k0 + 32, 0, 0);
    }
    ABfrag b00, b01, b10, b11;            // K^T fragments, shared by both m-tiles
    b00.v = *(const v16bf*)(Kp + (k0 + ln) * 64 + 16 * g);
    b01.v = *(const v16bf*)(Kp + (k0 + ln) * 64 + 32 + 16 * g);
    b10.v = *(const v16bf*)(Kp + (k0 + 16 + ln) * 64 + 16 * g);
    b11.v = *(const v16bf*)(Kp + (k0 + 16 + ln) * 64 + 32 + 16 * g);

    v8f s[2][2];
#pragma unroll
    for (int mt2 = 0; mt2 < 2; ++mt2) {
      s[mt2][0] = 0.f; s[mt2][1] = 0.f;
      s[mt2][0] = wmma_bf16(qf[mt2][0].v, b00.v, s[mt2][0]);
      s[mt2][0] = wmma_bf16(qf[mt2][1].v, b01.v, s[mt2][0]);
      s[mt2][1] = wmma_bf16(qf[mt2][0].v, b10.v, s[mt2][1]);
      s[mt2][1] = wmma_bf16(qf[mt2][1].v, b11.v, s[mt2][1]);
    }

    float fr[2][8];
#pragma unroll
    for (int mt2 = 0; mt2 < 2; ++mt2) {
#pragma unroll
      for (int r = 0; r < 8; ++r) {
        const int qi = qbase + mt2 * 16 + r + 8 * g;
        float a0 = s[mt2][0][r] * 0.125f;           // 1/sqrt(64)
        float a1 = s[mt2][1][r] * 0.125f;
        if (k0 + ln > qi)      a0 = -1e30f;         // causal mask
        if (k0 + 16 + ln > qi) a1 = -1e30f;
        float v = fmaxf(a0, a1);
#pragma unroll
        for (int off = 1; off < 16; off <<= 1) v = fmaxf(v, __shfl_xor(v, off));
        const float mn = fmaxf(m[mt2][r], v);
        fr[mt2][r] = __expf(m[mt2][r] - mn);
        m[mt2][r] = mn;
        const float p0 = __expf(a0 - mn);
        const float p1 = __expf(a1 - mn);
        float rs = p0 + p1;
#pragma unroll
        for (int off = 1; off < 16; off <<= 1) rs += __shfl_xor(rs, off);
        l[mt2][r] = l[mt2][r] * fr[mt2][r] + rs;
        const int row = r + 8 * g;
        pbuf[mt2 * 512 + row * 32 + ln]      = f2bf(p0);
        pbuf[mt2 * 512 + row * 32 + 16 + ln] = f2bf(p1);
      }
#pragma unroll
      for (int n = 0; n < 4; ++n)
#pragma unroll
        for (int r = 0; r < 8; ++r) acc[mt2][n][r] = acc[mt2][n][r] * fr[mt2][r];
    }

    asm volatile("" ::: "memory");        // LDS writes stay before reads
    ABfrag pf[2];
#pragma unroll
    for (int mt2 = 0; mt2 < 2; ++mt2) {
      pf[mt2].h.lo = *(const v8bf*)(pbuf + mt2 * 512 + ln * 32 + 16 * g);
      pf[mt2].h.hi = *(const v8bf*)(pbuf + mt2 * 512 + ln * 32 + 8 + 16 * g);
    }
#pragma unroll
    for (int n = 0; n < 4; ++n) {
      ABfrag vf;                          // V fragment shared by both m-tiles
      vf.v = *(const v16bf*)(Vp + (size_t)(n * 16 + ln) * 1024 + k0 + 16 * g);
      acc[0][n] = wmma_bf16(pf[0].v, vf.v, acc[0][n]);
      acc[1][n] = wmma_bf16(pf[1].v, vf.v, acc[1][n]);
    }
    asm volatile("" ::: "memory");        // next-iter LDS stores stay after reads
  }

  const int b = bh / 12, h = bh % 12;
#pragma unroll
  for (int mt2 = 0; mt2 < 2; ++mt2) {
#pragma unroll
    for (int n = 0; n < 4; ++n) {
#pragma unroll
      for (int r = 0; r < 8; ++r) {
        const int q = qbase + mt2 * 16 + r + 8 * g;
        ybf[((size_t)b * 1024 + q) * 768 + h * 64 + n * 16 + ln] =
            f2bf(acc[mt2][n][r] / l[mt2][r]);
      }
    }
  }
}

// ---------------- launcher ----------------
extern "C" void kernel_launch(void* const* d_in, const int* in_sizes, int n_in,
                              void* d_out, int out_size, void* d_ws, size_t ws_size,
                              hipStream_t stream) {
  const float* x      = (const float*)d_in[0];
  const float* W_attn = (const float*)d_in[1];
  const float* b_attn = (const float*)d_in[2];
  const float* W_proj = (const float*)d_in[3];
  const float* b_proj = (const float*)d_in[4];
  float* out = (float*)d_out;

  char* ws = (char*)d_ws;
  const size_t E = 16384ull * 768ull;
  bf16* xbf    = (bf16*)ws; ws += E * 2;
  bf16* WattnT = (bf16*)ws; ws += 2304ull * 768ull * 2;
  bf16* WprojT = (bf16*)ws; ws += 768ull * 768ull * 2;
  bf16* Qb     = (bf16*)ws; ws += E * 2;
  bf16* Kb     = (bf16*)ws; ws += E * 2;
  bf16* Vt     = (bf16*)ws; ws += E * 2;
  bf16* ybf    = (bf16*)ws;

  cvt_f32_bf16<<<(int)((E + 255) / 256), 256, 0, stream>>>(x, xbf, (int)E);
  transpose_to_bf16<<<(2304 * 768 + 255) / 256, 256, 0, stream>>>(W_attn, WattnT, 768, 2304);
  transpose_to_bf16<<<(768 * 768 + 255) / 256, 256, 0, stream>>>(W_proj, WprojT, 768, 768);

  // QKV: (16384/128) x (2304/128) = 128 x 18 blocks
  gemm_wmma<<<128 * 18, 256, 0, stream>>>(xbf, WattnT, b_attn, 2304, 0,
                                          Qb, Kb, Vt, nullptr);
  // Attention: 192 heads * 32 query blocks = 6144 waves, 4/block
  attn_wmma<<<1536, 128, 0, stream>>>(Qb, Kb, Vt, ybf);
  // Projection: (16384/128) x (768/128) = 128 x 6 blocks
  gemm_wmma<<<128 * 6, 256, 0, stream>>>(ybf, WprojT, b_proj, 768, 1,
                                         nullptr, nullptr, nullptr, out);
}
